// GraphiT_GT_Layer_56581899158014
// MI455X (gfx1250) — matrix operand order repaired and checked by reading, not compiled, not run on hardware
//
#include <hip/hip_runtime.h>
#include <math.h>

#define B_ 4
#define N_ 512
#define D_ 64
#define H_ 8
#define DH_ 8

typedef __attribute__((ext_vector_type(16))) __bf16 v16bf;
typedef __attribute__((ext_vector_type(8)))  float  v8f;

union Frag { v16bf v; unsigned int u[8]; };

__device__ __forceinline__ unsigned short f2bf(float f) {
  unsigned u = __float_as_uint(f);
  u += 0x7FFFu + ((u >> 16) & 1u);          // round-to-nearest-even bf16
  return (unsigned short)(u >> 16);
}
__device__ __forceinline__ unsigned pack2(float lo, float hi) {
  return (unsigned)f2bf(lo) | ((unsigned)f2bf(hi) << 16);
}

__device__ __forceinline__ float wred_sum(float v) {
#pragma unroll
  for (int o = 1; o < 32; o <<= 1) v += __shfl_xor(v, o, 32);
  return v;
}

__device__ __forceinline__ void atomicMaxF(float* a, float v) {
  if (v >= 0.f) atomicMax((int*)a, __float_as_int(v));
  else          atomicMin((unsigned int*)a, __float_as_uint(v));
}

// ---------------- Phase P: x = LN(h+p); Q,K*s,V,Q2,K2*s projections ---------
__global__ void __launch_bounds__(256) phaseP(
    const float* __restrict__ h, const float* __restrict__ p,
    const float* __restrict__ W_Q, const float* __restrict__ W_K,
    const float* __restrict__ W_V, const float* __restrict__ W_Q2,
    const float* __restrict__ W_K2,
    const float* __restrict__ ln1_g, const float* __restrict__ ln1_b,
    float* __restrict__ Qb, float* __restrict__ Kb,
    float* __restrict__ Q2b, float* __restrict__ K2b,
    float* __restrict__ Vb, float* __restrict__ maxb) {
  int lane = threadIdx.x & 31;
  int wave = threadIdx.x >> 5;
  if (blockIdx.x == 0 && threadIdx.x < B_ * H_)
    maxb[threadIdx.x] = -INFINITY;             // reset per launch (graph replay safe)
  int row = blockIdx.x * 8 + wave;             // b*N + i
  if (row >= B_ * N_) return;
  int c0 = lane, c1 = lane + 32;
  size_t ro = (size_t)row * D_;
  float x0 = h[ro + c0] + p[ro + c0];
  float x1 = h[ro + c1] + p[ro + c1];
  float m  = wred_sum(x0 + x1) * (1.f / 64.f);
  float va = wred_sum((x0 - m) * (x0 - m) + (x1 - m) * (x1 - m)) * (1.f / 64.f);
  float rs = rsqrtf(va + 1e-5f);
  x0 = (x0 - m) * rs * ln1_g[c0] + ln1_b[c0];
  x1 = (x1 - m) * rs * ln1_g[c1] + ln1_b[c1];
  const float scale = 0.35355339059327373f;    // DH^-0.5
  float q0 = 0, q1 = 0, k0 = 0, k1 = 0, v0 = 0, v1 = 0;
  float s0 = 0, s1 = 0, t0 = 0, t1 = 0;        // Q2, K2
  for (int d = 0; d < 64; ++d) {
    float xd = (d < 32) ? __shfl(x0, d, 32) : __shfl(x1, d - 32, 32);
    const float* wq  = W_Q  + d * 64;
    const float* wk  = W_K  + d * 64;
    const float* wv  = W_V  + d * 64;
    const float* wq2 = W_Q2 + d * 64;
    const float* wk2 = W_K2 + d * 64;
    q0 += xd * wq[c0];  q1 += xd * wq[c1];
    k0 += xd * wk[c0];  k1 += xd * wk[c1];
    v0 += xd * wv[c0];  v1 += xd * wv[c1];
    s0 += xd * wq2[c0]; s1 += xd * wq2[c1];
    t0 += xd * wk2[c0]; t1 += xd * wk2[c1];
  }
  Qb[ro + c0]  = q0;        Qb[ro + c1]  = q1;
  Kb[ro + c0]  = k0*scale;  Kb[ro + c1]  = k1*scale;
  Vb[ro + c0]  = v0;        Vb[ro + c1]  = v1;
  Q2b[ro + c0] = s0;        Q2b[ro + c1] = s1;
  K2b[ro + c0] = t0*scale;  K2b[ro + c1] = t1*scale;
}

// ---------------- Phase A: fused e@W_E(2) (bf16 WMMA) + score epilogue ------
__global__ void __launch_bounds__(128) phaseA(
    const float* __restrict__ e, const unsigned char* __restrict__ adj,
    const float* __restrict__ W_E, const float* __restrict__ W_E2,
    const float* __restrict__ Qb, const float* __restrict__ Kb,
    const float* __restrict__ Q2b, const float* __restrict__ K2b,
    float* __restrict__ scores, float* __restrict__ maxb) {
  int lane = threadIdx.x & 31;
  int wave = threadIdx.x >> 5;
  int bi   = blockIdx.x;                 // b*N + i
  int b    = bi >> 9;
  int i    = bi & (N_ - 1);
  size_t qo = (size_t)bi * 64;

  // B fragments: W_E columns pre-scaled by Q[b,i,:] (W_E2 by Q2). 32x16 bf16,
  // lane L = column n = L%16, u[g] = K-rows {k0, k0+1} in table order (8h+2g / 16+8h+2(g-4)).
  Frag WE[2][4], WF[2][4];
#pragma unroll
  for (int kb = 0; kb < 2; ++kb) {
#pragma unroll
    for (int c = 0; c < 4; ++c) {
      int n0 = c * 16 + (lane & 15);
#pragma unroll
      for (int g = 0; g < 8; ++g) {
        int d = kb * 32 + (g < 4 ? 8 * (lane >> 4) + 2 * g : 16 + 8 * (lane >> 4) + 2 * (g - 4));
        WE[kb][c].u[g] = pack2(W_E[d * 64 + n0]  * Qb[qo + n0],  W_E[(d + 1) * 64 + n0]  * Qb[qo + n0]);
        WF[kb][c].u[g] = pack2(W_E2[d * 64 + n0] * Q2b[qo + n0], W_E2[(d + 1) * 64 + n0] * Q2b[qo + n0]);
      }
    }
  }

  const unsigned char* adjp = adj + (size_t)bi * N_;
  int mrow = lane & 15, half = lane >> 4;
  float vmax[4] = {-INFINITY, -INFINITY, -INFINITY, -INFINITY};

  for (int t = 0; t < 8; ++t) {
    int jt = wave * 8 + t;

    // adjacency flags for this 16-row tile: 1 byte/lane + lane shuffles
    int ai = adjp[jt * 16 + mrow];
    unsigned long long bal = __ballot(ai != 0);
    bool anyE = (bal & 0xFFFFFFFFull) != 0ull;           // any adj row -> E path
    bool anyF = (bal & 0xFFFFFFFFull) != 0xFFFFFFFFull;  // any non-adj -> E2 path
    int avr[8];
#pragma unroll
    for (int r = 0; r < 8; ++r) avr[r] = __shfl(ai, r + (half << 3), 32);

    // A fragment: 16x32 bf16, lane holds row M = mrow. K runs {0..7} and
    // {16..23} (+8*half) are contiguous -> four float4 loads per K-block.
    const float4* er4 =
        (const float4*)(e + ((size_t)bi * N_ + (jt * 16 + mrow)) * 64);
    Frag a0, a1;
#pragma unroll
    for (int kb = 0; kb < 2; ++kb) {      // kb: K-offset 0 / 32
      Frag& af = kb ? a1 : a0;
      float4 lo0 = er4[kb * 8 + 2 * half];          // K = kb*32 + 8h .. +3
      float4 lo1 = er4[kb * 8 + 2 * half + 1];      // K = kb*32 + 8h+4 .. +7
      float4 hi0 = er4[kb * 8 + 4 + 2 * half];      // K = kb*32 + 16+8h ..
      float4 hi1 = er4[kb * 8 + 4 + 2 * half + 1];
      af.u[0] = pack2(lo0.x, lo0.y);
      af.u[1] = pack2(lo0.z, lo0.w);
      af.u[2] = pack2(lo1.x, lo1.y);
      af.u[3] = pack2(lo1.z, lo1.w);
      af.u[4] = pack2(hi0.x, hi0.y);
      af.u[5] = pack2(hi0.z, hi0.w);
      af.u[6] = pack2(hi1.x, hi1.y);
      af.u[7] = pack2(hi1.z, hi1.w);
    }

#pragma unroll
    for (int c = 0; c < 4; ++c) {
      v8f accE = {};
      v8f accF = {};
      if (anyE) {   // wave-uniform scalar branch: EXEC stays all-ones
        accE = __builtin_amdgcn_wmma_f32_16x16x32_bf16(false, a0.v, false, WE[0][c].v,
                                                       (short)0, accE, false, false);
        accE = __builtin_amdgcn_wmma_f32_16x16x32_bf16(false, a1.v, false, WE[1][c].v,
                                                       (short)0, accE, false, false);
      }
      if (anyF) {
        accF = __builtin_amdgcn_wmma_f32_16x16x32_bf16(false, a0.v, false, WF[0][c].v,
                                                       (short)0, accF, false, false);
        accF = __builtin_amdgcn_wmma_f32_16x16x32_bf16(false, a1.v, false, WF[1][c].v,
                                                       (short)0, accF, false, false);
      }
      // D layout: lane holds column n = c*16 + lane%16, rows m = r + 8*(lane>=16)
      int n  = c * 16 + (lane & 15);
      int hh = 2 * c + ((lane >> 3) & 1);
      float sval = 0.f;
#pragma unroll
      for (int r = 0; r < 8; ++r) {
        int j = jt * 16 + r + (half << 3);
        const float* kbase = avr[r] ? Kb : K2b;
        float pv = avr[r] ? accE[r] : accF[r];
        float kv = kbase[((size_t)(b * N_) + j) * 64 + n];
        float part = pv * kv;                  // Σ_k over 8 lanes of a head
        part += __shfl_xor(part, 1, 32);
        part += __shfl_xor(part, 2, 32);
        part += __shfl_xor(part, 4, 32);
        vmax[c] = fmaxf(vmax[c], part);
        if ((lane & 7) == r) sval = part;
      }
      int j = jt * 16 + (half << 3) + (lane & 7);
      scores[((size_t)(b * H_ + hh) * N_ + i) * N_ + j] = sval;
    }
  }
#pragma unroll
  for (int c = 0; c < 4; ++c) {
    float mx = fmaxf(vmax[c], __shfl_xor(vmax[c], 16, 32));
    if (lane == 0) atomicMaxF(&maxb[b * H_ + 2 * c], mx);
    if (lane == 8) atomicMaxF(&maxb[b * H_ + 2 * c + 1], mx);
  }
}

// ---------------- Phase B: exp/mask/k_RW weighting, att = scores@V/denom ----
__global__ void __launch_bounds__(256) phaseB(
    const float* __restrict__ scores, const float* __restrict__ maxb,
    const float* __restrict__ mask, const float* __restrict__ kRW,
    const float* __restrict__ Vb, float* __restrict__ hatt) {
  int lane = threadIdx.x & 31;
  int wave = threadIdx.x >> 5;
  int idx  = blockIdx.x * 8 + wave;      // ((b*H+h)*N + i)
  if (idx >= B_ * H_ * N_) return;
  int i  = idx & (N_ - 1);
  int bh = idx >> 9;
  int hh = bh & (H_ - 1);
  int b  = bh >> 3;
  float mx = maxb[bh];
  float mi = mask[b * N_ + i];
  const float* srow = scores + (size_t)idx * N_;
  const float* krw  = kRW + ((size_t)(b * N_) + i) * N_;
  const float* mk   = mask + b * N_;
  float den = 0.f;
  float acc[8] = {0, 0, 0, 0, 0, 0, 0, 0};
  for (int j = lane; j < N_; j += 32) {
    float w = __expf(srow[j] - mx) * mi * mk[j] * krw[j];
    den += w;
    const float* vr = Vb + ((size_t)(b * N_) + j) * 64 + hh * 8;
#pragma unroll
    for (int d = 0; d < 8; ++d) acc[d] += w * vr[d];
  }
  den = wred_sum(den);
#pragma unroll
  for (int d = 0; d < 8; ++d) acc[d] = wred_sum(acc[d]);
  float inv = 1.f / fmaxf(den, 1e-6f);
  if (lane == 0) {
    float* o = hatt + ((size_t)(b * N_) + i) * 64 + hh * 8;
#pragma unroll
    for (int d = 0; d < 8; ++d) o[d] = acc[d] * inv;
  }
}

// ---------------- Phase C: out-proj + residual + LN2 + FFN + residual -------
__global__ void __launch_bounds__(256) phaseC(
    const float* __restrict__ hatt, const float* __restrict__ h,
    const float* __restrict__ O_W, const float* __restrict__ O_b,
    const float* __restrict__ ln2_g, const float* __restrict__ ln2_b,
    const float* __restrict__ W_f1, const float* __restrict__ b_f1,
    const float* __restrict__ W_f2, const float* __restrict__ b_f2,
    float* __restrict__ out) {
  int lane = threadIdx.x & 31;
  int wave = threadIdx.x >> 5;
  int row  = blockIdx.x * 8 + wave;
  if (row >= B_ * N_) return;
  int c0 = lane, c1 = lane + 32;
  size_t ro = (size_t)row * 64;
  float a0 = hatt[ro + c0], a1 = hatt[ro + c1];
  float h0 = O_b[c0] + h[ro + c0];
  float h1 = O_b[c1] + h[ro + c1];
  for (int d = 0; d < 64; ++d) {
    float ad = (d < 32) ? __shfl(a0, d, 32) : __shfl(a1, d - 32, 32);
    h0 += ad * O_W[d * 64 + c0];
    h1 += ad * O_W[d * 64 + c1];
  }
  float m  = wred_sum(h0 + h1) * (1.f / 64.f);
  float va = wred_sum((h0 - m) * (h0 - m) + (h1 - m) * (h1 - m)) * (1.f / 64.f);
  float rs = rsqrtf(va + 1e-5f);
  float y0 = (h0 - m) * rs * ln2_g[c0] + ln2_b[c0];
  float y1 = (h1 - m) * rs * ln2_g[c1] + ln2_b[c1];
  float f[4];
#pragma unroll
  for (int t = 0; t < 4; ++t) f[t] = b_f1[lane + t * 32];
  for (int d = 0; d < 64; ++d) {
    float yd = (d < 32) ? __shfl(y0, d, 32) : __shfl(y1, d - 32, 32);
    const float* w1 = W_f1 + d * 128;
#pragma unroll
    for (int t = 0; t < 4; ++t) f[t] += yd * w1[lane + t * 32];
  }
#pragma unroll
  for (int t = 0; t < 4; ++t) f[t] = fmaxf(f[t], 0.f);
  float o0 = h0 + b_f2[c0], o1 = h1 + b_f2[c1];
#pragma unroll
  for (int t = 0; t < 4; ++t) {
    for (int l = 0; l < 32; ++l) {
      float fc = __shfl(f[t], l, 32);
      int c2 = t * 32 + l;
      o0 += fc * W_f2[c2 * 64 + c0];
      o1 += fc * W_f2[c2 * 64 + c1];
    }
  }
  out[ro + c0] = o0;
  out[ro + c1] = o1;
}

extern "C" void kernel_launch(void* const* d_in, const int* in_sizes, int n_in,
                              void* d_out, int out_size, void* d_ws, size_t ws_size,
                              hipStream_t stream) {
  (void)in_sizes; (void)n_in; (void)out_size; (void)ws_size;
  const float* h    = (const float*)d_in[0];
  const float* p    = (const float*)d_in[1];
  const float* e    = (const float*)d_in[2];
  const float* kRW  = (const float*)d_in[3];
  const float* mask = (const float*)d_in[4];
  const unsigned char* adj = (const unsigned char*)d_in[5];   // jax bool = 1 byte
  const float* W_Q  = (const float*)d_in[6];
  const float* W_K  = (const float*)d_in[7];
  const float* W_V  = (const float*)d_in[8];
  const float* W_Q2 = (const float*)d_in[9];
  const float* W_K2 = (const float*)d_in[10];
  const float* W_E  = (const float*)d_in[11];
  const float* W_E2 = (const float*)d_in[12];
  const float* O_W  = (const float*)d_in[13];
  const float* O_b  = (const float*)d_in[14];
  const float* ln1g = (const float*)d_in[15];
  const float* ln1b = (const float*)d_in[16];
  const float* ln2g = (const float*)d_in[17];
  const float* ln2b = (const float*)d_in[18];
  const float* W_f1 = (const float*)d_in[19];
  const float* b_f1 = (const float*)d_in[20];
  const float* W_f2 = (const float*)d_in[21];
  const float* b_f2 = (const float*)d_in[22];

  float* ws     = (float*)d_ws;
  float* scores = ws;                                  // B*H*N*N   = 8388608 f32
  float* Qb     = ws + (size_t)8388608;
  float* Kb     = Qb  + 131072;
  float* Q2b    = Kb  + 131072;
  float* K2b    = Q2b + 131072;
  float* Vb     = K2b + 131072;
  float* hatt   = Vb  + 131072;
  float* maxb   = hatt + 131072;                       // B*H = 32 f32
  float* out    = (float*)d_out;

  phaseP<<<256, 256, 0, stream>>>(h, p, W_Q, W_K, W_V, W_Q2, W_K2, ln1g, ln1b,
                                  Qb, Kb, Q2b, K2b, Vb, maxb);
  phaseA<<<B_ * N_, 128, 0, stream>>>(e, adj, W_E, W_E2, Qb, Kb, Q2b, K2b,
                                      scores, maxb);
  phaseB<<<2048, 256, 0, stream>>>(scores, maxb, mask, kRW, Vb, hatt);
  phaseC<<<256, 256, 0, stream>>>(hatt, h, O_W, O_b, ln2g, ln2b,
                                  W_f1, b_f1, W_f2, b_f2, out);
}
